// NLDGNN_50852412785073
// MI455X (gfx1250) — matrix operand order repaired and checked by reading, not compiled, not run on hardware
//
#include <hip/hip_runtime.h>
#include <hip/hip_bf16.h>

typedef __bf16 bf16x16 __attribute__((ext_vector_type(16)));
typedef float  v8f     __attribute__((ext_vector_type(8)));

__device__ __forceinline__ unsigned short f2bf(float f) {
    unsigned u = __float_as_uint(f);
    u += 0x7fffu + ((u >> 16) & 1u);     // RNE
    return (unsigned short)(u >> 16);
}

// ---------------------------------------------------------------------------
// WMMA NT GEMM: C[M,Nc] = A[M,K] * B[Nc,K]^T (+ bias per col).  bf16 inputs
// converted in-flight, fp32 accumulate via v_wmma_f32_16x16x32_bf16.
// Block tile 128x64, 8 waves (wave w -> rows 16w..16w+15, all 64 cols).
// Double-buffered LDS pipeline; branch-free float2 loads on full tiles.
// NOTE: requires K even (all call sites: 10000 / 500 / 256).
// ---------------------------------------------------------------------------
__device__ __forceinline__ void tile_load(
    const float* __restrict__ A, const float* __restrict__ B,
    int M, int Nc, int K, int rowBase, int colBase, int k0, int tid,
    float2 (&ar)[8], float2 (&br)[4])
{
    const bool fullA = (rowBase + 128 <= M) && (k0 + 32 <= K);
    const bool fullB = (colBase + 64 <= Nc) && (k0 + 32 <= K);
    #pragma unroll
    for (int it = 0; it < 8; ++it) {
        int q = tid + it * 256;           // 0..2047
        int r = q >> 4;                   // tile row
        int k = (q & 15) << 1;            // even k in [0,32)
        int gr = rowBase + r, gk = k0 + k;
        if (fullA) {
            ar[it] = *reinterpret_cast<const float2*>(&A[(long)gr * K + gk]);
        } else {
            float a0 = (gr < M && gk < K)     ? A[(long)gr * K + gk]     : 0.f;
            float a1 = (gr < M && gk + 1 < K) ? A[(long)gr * K + gk + 1] : 0.f;
            ar[it] = make_float2(a0, a1);
        }
    }
    #pragma unroll
    for (int it = 0; it < 4; ++it) {
        int q = tid + it * 256;           // 0..1023
        int n = q >> 4;
        int k = (q & 15) << 1;
        int gn = colBase + n, gk = k0 + k;
        if (fullB) {
            br[it] = *reinterpret_cast<const float2*>(&B[(long)gn * K + gk]);
        } else {
            float b0 = (gn < Nc && gk < K)     ? B[(long)gn * K + gk]     : 0.f;
            float b1 = (gn < Nc && gk + 1 < K) ? B[(long)gn * K + gk + 1] : 0.f;
            br[it] = make_float2(b0, b1);
        }
    }
}

__device__ __forceinline__ void tile_store(
    unsigned* As, unsigned* Bs, int tid,
    const float2 (&ar)[8], const float2 (&br)[4])
{
    #pragma unroll
    for (int it = 0; it < 8; ++it) {
        int q = tid + it * 256;
        int r = q >> 4;
        int k = (q & 15) << 1;
        // ISA A-layout (16-bit 16x32): lane = m + 16*k[3]; p = 4*(k>=16)+(k&7)/2
        int ln = (r & 15) + (((k >> 3) & 1) << 4);
        int p  = ((k & 16) >> 2) + ((k & 7) >> 1);
        As[((r >> 4) * 32 + ln) * 8 + p] =
            ((unsigned)f2bf(ar[it].y) << 16) | (unsigned)f2bf(ar[it].x);
    }
    #pragma unroll
    for (int it = 0; it < 4; ++it) {
        int q = tid + it * 256;
        int n = q >> 4;
        int k = (q & 15) << 1;
        // B operand (K x N): lane = n + 16*k[3]; p = 4*(k>=16)+(k&7)/2
        int ln = (n & 15) + (((k >> 3) & 1) << 4);
        int p  = ((k & 16) >> 2) + ((k & 7) >> 1);
        Bs[((n >> 4) * 32 + ln) * 8 + p] =
            ((unsigned)f2bf(br[it].y) << 16) | (unsigned)f2bf(br[it].x);
    }
}

__global__ __launch_bounds__(256) void gemm_nt_wmma(
    const float* __restrict__ A, const float* __restrict__ B,
    const float* __restrict__ bias, float* __restrict__ C,
    int M, int Nc, int K)
{
    __shared__ alignas(32) unsigned As[2][2048];   // 2 x 8 KB
    __shared__ alignas(32) unsigned Bs[2][1024];   // 2 x 4 KB
    const int tid  = threadIdx.x;
    const int wave = tid >> 5, lane = tid & 31;
    const int rowBase = blockIdx.x * 128;
    const int colBase = blockIdx.y * 64;

    v8f acc[4] = {};
    float2 ar[8];
    float2 br[4];

    tile_load(A, B, M, Nc, K, rowBase, colBase, 0, tid, ar, br);
    tile_store(As[0], Bs[0], tid, ar, br);

    int buf = 0;
    for (int k0 = 0; k0 < K; k0 += 32) {
        __syncthreads();
        const bool hasNext = (k0 + 32 < K);
        if (hasNext)
            tile_load(A, B, M, Nc, K, rowBase, colBase, k0 + 32, tid, ar, br);

        bf16x16 av = *reinterpret_cast<const bf16x16*>(
            &As[buf][(wave * 32 + lane) * 8]);
        #pragma unroll
        for (int t = 0; t < 4; ++t) {
            bf16x16 bv = *reinterpret_cast<const bf16x16*>(
                &Bs[buf][(t * 32 + lane) * 8]);
            acc[t] = __builtin_amdgcn_wmma_f32_16x16x32_bf16(
                false, av, false, bv, (short)0, acc[t], false, false);
        }

        if (hasNext)
            tile_store(As[buf ^ 1], Bs[buf ^ 1], tid, ar, br);
        buf ^= 1;
    }
    // ---- store: VGPR r of C holds row r (lanes 0-15) / r+8 (lanes 16-31) ----
    #pragma unroll
    for (int t = 0; t < 4; ++t) {
        int gc = colBase + t * 16 + (lane & 15);
        if (gc >= Nc) continue;
        float bv = bias ? bias[gc] : 0.f;
        #pragma unroll
        for (int r = 0; r < 8; ++r) {
            int gr = rowBase + wave * 16 + r + ((lane >> 4) << 3);
            if (gr < M) C[(long)gr * Nc + gc] = acc[t][r] + bv;
        }
    }
}

// ---------------------------------------------------------------------------
// Elementwise / small kernels
// ---------------------------------------------------------------------------
__global__ void zero_kernel(float* p, long n) {
    long i = (long)blockIdx.x * blockDim.x + threadIdx.x;
    if (i < n) p[i] = 0.f;
}

__global__ void fuse_relu(float* RAW, const float* XA, long n) {
    long i = (long)blockIdx.x * blockDim.x + threadIdx.x;
    if (i < n) RAW[i] = fmaxf(0.5f * RAW[i] + 0.5f * XA[i], 0.f);
}

__global__ void argmax16(const float* H, int* yhat, int n) {
    int i = blockIdx.x * blockDim.x + threadIdx.x;
    if (i >= n) return;
    const float* h = H + (long)i * 16;
    int best = 0; float bv = h[0];
    for (int c = 1; c < 16; ++c) { float v = h[c]; if (v > bv) { bv = v; best = c; } }
    yhat[i] = best;
}

__global__ void nei_scatter(float* nei, const float* aw, const int* ra,
                            const int* ca, const int* yhat, int E) {
    int e = blockIdx.x * blockDim.x + threadIdx.x;
    if (e >= E) return;
    atomicAdd(&nei[(long)ra[e] * 16 + yhat[ca[e]]], aw[e]);
}

__global__ void nei_mix(float* H, const float* nei, const float* lnw,
                        const float* lnb, int n) {
    int i = blockIdx.x * blockDim.x + threadIdx.x;
    if (i >= n) return;
    float nn[16]; float s = 0.f;
    for (int c = 0; c < 16; ++c) { nn[c] = nei[(long)i * 16 + c]; s += nn[c]; }
    float inv = (s != 0.f) ? 1.f / s : 0.f;
    for (int c = 0; c < 16; ++c) nn[c] *= inv;
    for (int c = 0; c < 16; ++c) {
        float a = lnb[c];
        for (int d = 0; d < 16; ++d) a += nn[d] * lnw[c * 16 + d];
        a = fmaxf(a, 0.f);
        H[(long)i * 16 + c] = 0.5f * H[(long)i * 16 + c] + 0.5f * a;  // W1 = 0.5
    }
}

__global__ void lx_kernel(float* LX, const float* H, const float* lw, int n) {
    int i = blockIdx.x * blockDim.x + threadIdx.x;
    if (i >= n) return;
    float hr[16];
    for (int c = 0; c < 16; ++c) hr[c] = H[(long)i * 16 + c];
    for (int c = 0; c < 16; ++c) {
        float a = 0.f;
        for (int d = 0; d < 16; ++d) a += hr[d] * lw[c * 16 + d];
        LX[(long)i * 16 + c] = a;
    }
}

__global__ void conf_kernel(float* conf, const float* LX, const int* ra,
                            const int* ca, const float* attw, const float* cp,
                            const float* cin, int E) {
    int e = blockIdx.x * blockDim.x + threadIdx.x;
    if (e >= E) return;
    const float* lr = LX + (long)ra[e] * 16;
    const float* lc = LX + (long)ca[e] * 16;
    float a = 0.f;
    for (int c = 0; c < 16; ++c) a += fmaxf(lr[c] + lc[c], 0.f) * attw[c];
    float sg = 1.f / (1.f + expf(-a));
    float m  = fmaxf(cp[0], fmaxf(cp[1], cp[2]));
    float e0 = expf(cp[0] - m), e1 = expf(cp[1] - m), e2 = expf(cp[2] - m);
    float is = 1.f / (e0 + e1 + e2);
    conf[e] = sg * e0 * is + cin[e] * e1 * is + cin[(long)E + e] * e2 * is;
}

// ---------------------------------------------------------------------------
// Edge residual: s[e] = w[e] - dot(P[row[e]], Q[col[e]])  (K=64, wave/edge)
// ---------------------------------------------------------------------------
__global__ __launch_bounds__(256) void edge_score(
    float* sOut, const float* ew, const int* ra, const int* ca,
    const float* P, const float* Q, int E)
{
    int wave = threadIdx.x >> 5, lane = threadIdx.x & 31;
    int e = blockIdx.x * 8 + wave;
    if (e >= E) return;
    long r = ra[e], c = ca[e];
    float a = P[r * 64 + lane] * Q[c * 64 + lane]
            + P[r * 64 + 32 + lane] * Q[c * 64 + 32 + lane];
    for (int off = 16; off; off >>= 1) a += __shfl_xor(a, off, 32);
    if (lane == 0) sOut[e] = ew[e] - a;
}

// ---------------------------------------------------------------------------
// Quantile selection (q1=ss[E/4], q3=ss[3E/4] of S=s^2) via 3-level histogram
// ---------------------------------------------------------------------------
__global__ void qinit(unsigned* mm, unsigned* hist) {
    int t = threadIdx.x;
    if (t == 0) { mm[0] = 0x7f800000u; mm[1] = 0u; }
    for (int i = t; i < 2048; i += blockDim.x) hist[i] = 0u;
}

__global__ void qminmax(const float* s, int n, unsigned* mm) {
    __shared__ unsigned smin, smax;
    if (threadIdx.x == 0) { smin = 0x7f800000u; smax = 0u; }
    __syncthreads();
    unsigned lmin = 0x7f800000u, lmax = 0u;
    for (int i = blockIdx.x * blockDim.x + threadIdx.x; i < n;
         i += gridDim.x * blockDim.x) {
        float v = s[i]; v = v * v;
        unsigned b = __float_as_uint(v);       // v >= 0: uint order == float order
        lmin = (b < lmin) ? b : lmin;
        lmax = (b > lmax) ? b : lmax;
    }
    atomicMin(&smin, lmin); atomicMax(&smax, lmax);
    __syncthreads();
    if (threadIdx.x == 0) { atomicMin(&mm[0], smin); atomicMax(&mm[1], smax); }
}

__global__ void qsetup(float* qr, int* rk, const unsigned* mm, int r1, int r2) {
    float lo = __uint_as_float(mm[0]);
    float hi = __uint_as_float(mm[1]);
    hi = hi * 1.0000005f + 1e-30f;
    if (!(hi > lo)) hi = lo + 1e-30f;
    qr[0] = lo; qr[1] = hi; qr[2] = lo; qr[3] = hi;
    rk[0] = r1; rk[1] = r2;
}

__global__ void qhist(const float* s, int n, const float* qr, unsigned* hist) {
    __shared__ unsigned h[2048];
    for (int i = threadIdx.x; i < 2048; i += blockDim.x) h[i] = 0u;
    __syncthreads();
    float lo1 = qr[0], hi1 = qr[1], lo2 = qr[2], hi2 = qr[3];
    float iw1 = (hi1 > lo1) ? 1024.f / (hi1 - lo1) : 0.f;
    float iw2 = (hi2 > lo2) ? 1024.f / (hi2 - lo2) : 0.f;
    for (int i = blockIdx.x * blockDim.x + threadIdx.x; i < n;
         i += gridDim.x * blockDim.x) {
        float v = s[i]; float S = v * v;
        if (S >= lo1 && S < hi1) {
            int b = (int)((S - lo1) * iw1); b = b > 1023 ? 1023 : (b < 0 ? 0 : b);
            atomicAdd(&h[b], 1u);
        }
        if (S >= lo2 && S < hi2) {
            int b = (int)((S - lo2) * iw2); b = b > 1023 ? 1023 : (b < 0 ? 0 : b);
            atomicAdd(&h[1024 + b], 1u);
        }
    }
    __syncthreads();
    for (int i = threadIdx.x; i < 2048; i += blockDim.x)
        if (h[i]) atomicAdd(&hist[i], h[i]);
}

__global__ void qnarrow(float* qr, int* rk, unsigned* hist, int last) {
    if (threadIdx.x == 0) {
        for (int t = 0; t < 2; ++t) {
            float lo = qr[t * 2], hi = qr[t * 2 + 1];
            float w = (hi - lo) * (1.f / 1024.f);
            int r = rk[t];
            unsigned cum = 0; int b = 0;
            for (; b < 1024; ++b) {
                unsigned c = hist[t * 1024 + b];
                if (cum + c > (unsigned)r) break;
                cum += c;
            }
            if (b >= 1024) b = 1023;
            rk[t] = r - (int)cum;
            float nlo = lo + b * w;
            qr[t * 2] = nlo; qr[t * 2 + 1] = nlo + w;
            if (last) qr[4 + t] = nlo;     // qr[4]=q1, qr[5]=q3
        }
    }
    __syncthreads();
    for (int i = threadIdx.x; i < 2048; i += blockDim.x) hist[i] = 0u;
}

__global__ void mask_ss(float* SS, const float* s, const float* conf,
                        const float* qr, const float* bp, int E) {
    int e = blockIdx.x * blockDim.x + threadIdx.x;
    if (e >= E) return;
    float q1 = qr[4], q3 = qr[5];
    float thr = q3 + 1.5f * (q3 - q1);
    float sv = s[e];
    SS[e] = (sv * sv < thr * conf[e]) ? (sv + bp[0]) : 0.f;
}

// out[oIdx[e],:] += vals[e] * Gm[gIdx[e],:]   (K=64, thread per (e,k))
__global__ void spmm64(float* out, const float* vals, const int* oIdx,
                       const int* gIdx, const float* Gm, int E) {
    long gt = (long)blockIdx.x * blockDim.x + threadIdx.x;
    if (gt >= (long)E * 64) return;
    int e = (int)(gt >> 6), k = (int)(gt & 63);
    float v = vals[e];
    if (v != 0.f)
        atomicAdd(&out[(long)oIdx[e] * 64 + k], v * Gm[(long)gIdx[e] * 64 + k]);
}

// G[64,64] += X^T X over 128-row block
__global__ __launch_bounds__(256) void gram64(const float* X, int n, float* G) {
    __shared__ float rows[8][64];
    float acc[16];
    #pragma unroll
    for (int u = 0; u < 16; ++u) acc[u] = 0.f;
    int base = blockIdx.x * 128;
    for (int i0 = 0; i0 < 128; i0 += 8) {
        for (int t = threadIdx.x; t < 512; t += 256) {
            int rr = t >> 6, cc = t & 63; int gi = base + i0 + rr;
            rows[rr][cc] = (gi < n) ? X[(long)gi * 64 + cc] : 0.f;
        }
        __syncthreads();
        for (int rr = 0; rr < 8; ++rr) {
            #pragma unroll
            for (int u = 0; u < 16; ++u) {
                int idx = threadIdx.x + u * 256;
                acc[u] += rows[rr][idx >> 6] * rows[rr][idx & 63];
            }
        }
        __syncthreads();
    }
    #pragma unroll
    for (int u = 0; u < 16; ++u) atomicAdd(&G[threadIdx.x + u * 256], acc[u]);
}

// out[16,64] += H^T X over 128-row block
__global__ __launch_bounds__(256) void ht16x64(const float* H, const float* X,
                                               int n, float* out) {
    __shared__ float hs[8][16];
    __shared__ float xs[8][64];
    float acc[4] = {0.f, 0.f, 0.f, 0.f};
    int base = blockIdx.x * 128;
    for (int i0 = 0; i0 < 128; i0 += 8) {
        for (int t = threadIdx.x; t < 128; t += 256) {
            int rr = t >> 4, cc = t & 15; int gi = base + i0 + rr;
            hs[rr][cc] = (gi < n) ? H[(long)gi * 16 + cc] : 0.f;
        }
        for (int t = threadIdx.x; t < 512; t += 256) {
            int rr = t >> 6, cc = t & 63; int gi = base + i0 + rr;
            xs[rr][cc] = (gi < n) ? X[(long)gi * 64 + cc] : 0.f;
        }
        __syncthreads();
        for (int rr = 0; rr < 8; ++rr) {
            #pragma unroll
            for (int u = 0; u < 4; ++u) {
                int idx = threadIdx.x + u * 256;
                acc[u] += hs[rr][idx >> 6] * xs[rr][idx & 63];
            }
        }
        __syncthreads();
    }
    #pragma unroll
    for (int u = 0; u < 4; ++u) atomicAdd(&out[threadIdx.x + u * 256], acc[u]);
}

// INV = inv(scale*G + beta*I)   (64x64 Gauss-Jordan; SPD -> no pivoting)
__global__ __launch_bounds__(64) void inv64(const float* G, float scale,
                                            float beta, float* INV) {
    __shared__ float M[64][130];
    int r = threadIdx.x;
    for (int j = 0; j < 64; ++j) {
        M[r][j]      = scale * G[r * 64 + j] + ((r == j) ? beta : 0.f);
        M[r][64 + j] = (r == j) ? 1.f : 0.f;
    }
    __syncthreads();
    for (int k = 0; k < 64; ++k) {
        if (r == k) {
            float pv = M[k][k];
            pv = (pv != 0.f) ? 1.f / pv : 0.f;
            for (int j = 0; j < 128; ++j) M[k][j] *= pv;
        }
        __syncthreads();
        if (r != k) {
            float f = M[r][k];
            if (f != 0.f)
                for (int j = 0; j < 128; ++j) M[r][j] -= f * M[k][j];
        }
        __syncthreads();
    }
    for (int j = 0; j < 64; ++j) INV[r * 64 + j] = M[r][64 + j];
}

// Pnew[i,:] = (temp[i,:] + [Pold[i,:]@G + eps*h[i,:]@HtX]) @ INV
__global__ __launch_bounds__(64) void update_row(
    float* Pnew, const float* temp, const float* Pold, const float* G,
    const float* H, const float* HtX, const float* INV, float eps,
    int useExtra, int n)
{
    __shared__ float t[64], urow[64], hrow[16];
    int i = blockIdx.x;
    int j = threadIdx.x;
    if (useExtra) {
        urow[j] = Pold[(long)i * 64 + j];
        if (j < 16) hrow[j] = H[(long)i * 16 + j];
    }
    __syncthreads();
    float v = temp[(long)i * 64 + j];
    if (useExtra) {
        float g = 0.f;
        for (int m = 0; m < 64; ++m) g += urow[m] * G[m * 64 + j];
        float hh = 0.f;
        for (int c = 0; c < 16; ++c) hh += hrow[c] * HtX[c * 64 + j];
        v += g + eps * hh;
    }
    t[j] = v;
    __syncthreads();
    float o = 0.f;
    for (int k = 0; k < 64; ++k) o += t[k] * INV[k * 64 + j];
    Pnew[(long)i * 64 + j] = o;
}

// out = ((1-a)*U@HtV^T + a*H) @ W
__global__ void final_out(float* out, const float* U, const float* HtV,
                          const float* H, const float* W, int n) {
    int i = blockIdx.x * blockDim.x + threadIdx.x;
    if (i >= n) return;
    float o[16];
    const float* ur = U + (long)i * 64;
    for (int c = 0; c < 16; ++c) {
        float p = 0.f;
        const float* hv = HtV + c * 64;
        for (int k = 0; k < 64; ++k) p += ur[k] * hv[k];
        o[c] = 0.9f * p + 0.1f * H[(long)i * 16 + c];    // ALPHA = 0.1
    }
    for (int j = 0; j < 16; ++j) {
        float a = 0.f;
        for (int c = 0; c < 16; ++c) a += o[c] * W[c * 16 + j];
        out[(long)i * 16 + j] = a;
    }
}

// ---------------------------------------------------------------------------
extern "C" void kernel_launch(void* const* d_in, const int* in_sizes, int n_in,
                              void* d_out, int out_size, void* d_ws, size_t ws_size,
                              hipStream_t stream) {
    (void)in_sizes; (void)n_in; (void)out_size; (void)ws_size;
    const int Nn = 10000, Ee = 320000, NF = 500, NH = 256;

    const float* x    = (const float*)d_in[0];
    const int*   ei   = (const int*)d_in[1];
    const int*   row  = ei;
    const int*   col  = ei + Ee;
    const float* ew   = (const float*)d_in[2];
    const float* aw   = (const float*)d_in[3];
    const float* conn = (const float*)d_in[4];
    const float* cin  = (const float*)d_in[5];
    const float* l1w  = (const float*)d_in[6];
    const float* l1b  = (const float*)d_in[7];
    const float* l2w  = (const float*)d_in[8];
    const float* l2b  = (const float*)d_in[9];
    const float* law  = (const float*)d_in[10];
    const float* lab  = (const float*)d_in[11];
    const float* lnw  = (const float*)d_in[12];
    const float* lnb  = (const float*)d_in[13];
    const float* lw   = (const float*)d_in[14];
    const float* attw = (const float*)d_in[15];
    const float* vw   = (const float*)d_in[16];
    const float* vb   = (const float*)d_in[17];
    const float* Wm   = (const float*)d_in[18];
    const float* cp   = (const float*)d_in[19];
    const float* bp   = (const float*)d_in[20];
    float* out = (float*)d_out;

    // ---- workspace arena (floats) ----
    float* ws   = (float*)d_ws;
    float* RAW  = ws;                         // N*256
    float* SCR  = ws + 2560000;               // N*256 reuse region
    float* XA   = SCR;                        // transient (dead after fuse)
    float* CONF = SCR;                        // E
    float* SV   = SCR + 320000;               // E
    float* SS   = SCR + 640000;               // E
    float* TMP  = SCR + 960000;               // N*64
    float* LXb  = SCR + 1600000;              // N*16
    float* NEI  = SCR + 1760000;              // N*16
    int*   YH   = (int*)(SCR + 1920000);      // N
    float* H    = ws + 5120000;               // N*16
    float* Vb   = ws + 5280000;               // N*64
    float* Ub   = ws + 5920000;
    float* U2   = ws + 6560000;
    float* V2   = ws + 7200000;
    float* HtX  = ws + 7840000;               // 16*64
    float* G    = ws + 7841024;               // 64*64
    float* INVm = ws + 7845120;               // 64*64
    float* QR   = ws + 7849216;               // 8 floats
    int*   RK   = (int*)(ws + 7849224);       // 2 ints
    unsigned* MM   = (unsigned*)(QR + 6);     // min/max bits
    unsigned* HIST = (unsigned*)(ws + 7849232); // 2048

    dim3 blk(256);
    const int gM = (Nn + 127) / 128;

    // ---- h = relu((1-l)*(x@l1^T+b1) + l*(connect@law^T+lab)), l=0.5 ----
    gemm_nt_wmma<<<dim3(gM, 4), blk, 0, stream>>>(conn, law, lab, XA, Nn, NH, Nn);
    gemm_nt_wmma<<<dim3(gM, 4), blk, 0, stream>>>(x,    l1w, l1b, RAW, Nn, NH, NF);
    fuse_relu<<<(2560000 + 255) / 256, blk, 0, stream>>>(RAW, XA, 2560000L);

    // ---- h2 = raw@l2^T + b2 ; argmax ; neighbor label mix ----
    gemm_nt_wmma<<<dim3(gM, 1), blk, 0, stream>>>(RAW, l2w, l2b, H, Nn, 16, NH);
    argmax16<<<(Nn + 255) / 256, blk, 0, stream>>>(H, YH, Nn);
    zero_kernel<<<(160000 + 255) / 256, blk, 0, stream>>>(NEI, 160000L);
    nei_scatter<<<(Ee + 255) / 256, blk, 0, stream>>>(NEI, aw, row, col, YH, Ee);
    nei_mix<<<(Nn + 255) / 256, blk, 0, stream>>>(H, NEI, lnw, lnb, Nn);

    // ---- per-edge confidence ----
    lx_kernel<<<(Nn + 255) / 256, blk, 0, stream>>>(LXb, H, lw, Nn);
    conf_kernel<<<(Ee + 255) / 256, blk, 0, stream>>>(CONF, LXb, row, col, attw,
                                                      cp, cin, Ee);

    // ---- V = raw@vw^T + vb ; U = spmm(aw,row,V[col]) @ inv(V^T V) ----
    gemm_nt_wmma<<<dim3(gM, 1), blk, 0, stream>>>(RAW, vw, vb, Vb, Nn, 64, NH);
    zero_kernel<<<16, blk, 0, stream>>>(G, 4096L);
    gram64<<<gM, blk, 0, stream>>>(Vb, Nn, G);
    inv64<<<1, 64, 0, stream>>>(G, 1.0f, 0.0f, INVm);
    zero_kernel<<<2500, blk, 0, stream>>>(TMP, 640000L);
    spmm64<<<(int)(((long)Ee * 64 + 255) / 256), blk, 0, stream>>>(TMP, aw, row,
                                                                   col, Vb, Ee);
    update_row<<<Nn, 64, 0, stream>>>(Ub, TMP, nullptr, nullptr, nullptr,
                                      nullptr, INVm, 0.f, 0, Nn);

    // ---- quantile helper ----
    auto quantile = [&](const float* sArr) {
        qinit<<<1, blk, 0, stream>>>(MM, HIST);
        qminmax<<<512, blk, 0, stream>>>(sArr, Ee, MM);
        qsetup<<<1, 1, 0, stream>>>(QR, RK, MM, Ee / 4, 3 * (Ee / 4));
        for (int lev = 0; lev < 3; ++lev) {
            qhist<<<512, blk, 0, stream>>>(sArr, Ee, QR, HIST);
            qnarrow<<<1, blk, 0, stream>>>(QR, RK, HIST, lev == 2 ? 1 : 0);
        }
    };

    // ---- one U-or-V update: score -> quantile mask -> spmm -> solve ----
    auto do_update = [&](const float* P, const float* Q, float* Pnew,
                         const int* oIdx, const int* gIdx) {
        edge_score<<<(Ee + 7) / 8, blk, 0, stream>>>(SV, ew, row, col, P, Q, Ee);
        quantile(SV);
        mask_ss<<<(Ee + 255) / 256, blk, 0, stream>>>(SS, SV, CONF, QR, bp, Ee);
        zero_kernel<<<2500, blk, 0, stream>>>(TMP, 640000L);
        spmm64<<<(int)(((long)Ee * 64 + 255) / 256), blk, 0, stream>>>(TMP, SS,
                                                             oIdx, gIdx, Q, Ee);
        zero_kernel<<<16, blk, 0, stream>>>(G, 4096L);
        gram64<<<gM, blk, 0, stream>>>(Q, Nn, G);
        zero_kernel<<<4, blk, 0, stream>>>(HtX, 1024L);
        ht16x64<<<gM, blk, 0, stream>>>(H, Q, Nn, HtX);
        inv64<<<1, 64, 0, stream>>>(G, 1.1f, 1.0f, INVm);   // (1+EPS), BETA
        update_row<<<Nn, 64, 0, stream>>>(Pnew, TMP, P, G, H, HtX, INVm,
                                          0.1f, 1, Nn);     // EPS
    };

    // GAMMA = 4: u, v, u, v (explicit buffer schedule)
    do_update(Ub, Vb, U2, row, col);   // j=0: U' from (U,V)
    do_update(Vb, U2, V2, col, row);   // j=1: V' from (V,U')
    do_update(U2, V2, Ub, row, col);   // j=2
    do_update(V2, Ub, Vb, col, row);   // j=3  -> final U=Ub, V=Vb

    // ---- out = ((1-a)*U@(h^T V)^T + a*h) @ W ----
    zero_kernel<<<4, blk, 0, stream>>>(HtX, 1024L);
    ht16x64<<<gM, blk, 0, stream>>>(H, Vb, Nn, HtX);
    final_out<<<(Nn + 255) / 256, blk, 0, stream>>>(out, Ub, HtX, H, Wm, Nn);
}
